// Tc_Radar2D_9792525435690
// MI455X (gfx1250) — compile-verified, matrix operand order checked by reading, not run on hardware
//
#include <hip/hip_runtime.h>

// ---------------------------------------------------------------------------
// Types for CDNA5 WMMA (wave32): D(f32 16x16) = A(bf16 16x32) * B(bf16 32x16) + C
// ---------------------------------------------------------------------------
typedef __attribute__((ext_vector_type(16))) __bf16 v16bf;
typedef __attribute__((ext_vector_type(8)))  __bf16 v8bf;
typedef __attribute__((ext_vector_type(8)))  float  v8f;
typedef __attribute__((ext_vector_type(16))) short  v16s;

#define DEV __device__ __forceinline__

// ---------------------------------------------------------------------------
// Problem constants
// ---------------------------------------------------------------------------
constexpr size_t TOK   = 200704;          // 16 * 112 * 112 tokens after merge
constexpr int    DIMC  = 96;
constexpr int    NWINS = 4096;            // 16 batches * 16*16 windows
constexpr int    NWH   = NWINS * 3;       // window-heads
constexpr float  QSCALE = 0.17677669529663689f;   // 32^-0.5

// Workspace layout (bytes)
constexpr size_t OFF_XP = 0;                          // fp32 residual, TOK*96
constexpr size_t OFF_Y  = OFF_XP + TOK*96*4;          // bf16 activation / attn-out
constexpr size_t SZQKV  = (size_t)NWH*64*32*2;        // one padded window tensor
constexpr size_t OFF_Q  = OFF_Y + TOK*96*2;
constexpr size_t OFF_K  = OFF_Q + SZQKV;
constexpr size_t OFF_V  = OFF_K + SZQKV;
constexpr size_t OFF_FF = OFF_Q;                      // reuse q/k/v region for MLP hidden
constexpr size_t END_V  = OFF_V + SZQKV;
constexpr size_t END_FF = OFF_FF + TOK*384*2;
constexpr size_t OFF_W  = (END_V > END_FF ? END_V : END_FF);   // bf16 weight frags

// weight-fragment element offsets (in bf16 elements from OFF_W)
constexpr size_t WOFF_QKV = 0;          // 2 x (18 nt * 3 kt * 512) = 2*27648
constexpr size_t WOFF_OUT = 55296;      // 2 x ( 6 nt * 3 kt * 512) = 2*9216
constexpr size_t WOFF_FF1 = 73728;      // 2 x (24 nt * 3 kt * 512) = 2*36864
constexpr size_t WOFF_FF2 = 147456;     // 2 x ( 6 nt * 12kt * 512) = 2*36864

// ---------------------------------------------------------------------------
// Helpers
// ---------------------------------------------------------------------------
DEV __bf16 f2bf(float f) {                       // round-to-nearest-even
  unsigned u = __float_as_uint(f);
  unsigned r = (u + 0x7FFFu + ((u >> 16) & 1u)) >> 16;
  unsigned short s = (unsigned short)r;
  return __builtin_bit_cast(__bf16, s);
}

DEV v16bf zfrag() {
  v16s z = (v16s)0;
  return __builtin_bit_cast(v16bf, z);
}

// Build one lane's A/B fragment from two contiguous 8-element (16B) chunks.
DEV v16bf frag_pair(const __bf16* p0, const __bf16* p1) {
  v8bf lo = *(const v8bf*)p0;
  v8bf hi = *(const v8bf*)p1;
  v16bf r;
#pragma unroll
  for (int i = 0; i < 8; ++i) { r[i] = lo[i]; r[i+8] = hi[i]; }
  return r;
}

// Row-major source (contiguous K per row): ISA 16-bit A layout means lane needs
// K = kt*32 + g*8 + [0..7] and kt*32 + 16 + g*8 + [0..7].
DEV v16bf frag_row(const __bf16* rowptr, int kt, int g) {
  const __bf16* p = rowptr + kt*32 + g*8;
  return frag_pair(p, p + 16);
}

// Pre-swizzled weight fragments: [nt][kt][lane][16] contiguous.
DEV v16bf frag_w(const __bf16* w, int nt, int kt, int KT, int lane) {
  const __bf16* p = w + (((size_t)nt*KT + kt)*32 + (size_t)lane)*16;
  return frag_pair(p, p + 8);
}

DEV v8f wmma_bf(v16bf a, v16bf b, v8f c) {
  return __builtin_amdgcn_wmma_f32_16x16x32_bf16(false, a, false, b, (short)0, c,
                                                 false, false);
}

DEV float gelu_exact(float x) {
  return 0.5f * x * (1.0f + erff(x * 0.70710678118654752f));
}

// ---------------------------------------------------------------------------
// K0: fp32 row-major (KxN) weights -> bf16 WMMA B-fragment layout
// ---------------------------------------------------------------------------
__global__ __launch_bounds__(256) void k_cvt(const float* __restrict__ src,
                                             __bf16* __restrict__ dst,
                                             int K, int N) {
  int KT = K >> 5, NT = N >> 4;
  int idx = blockIdx.x * 256 + threadIdx.x;
  if (idx >= KT * NT * 512) return;
  int i = idx & 15, lane = (idx >> 4) & 31, f = idx >> 9;
  int kt = f % KT, nt = f / KT;
  int k = kt*32 + ((i < 8) ? i : i + 8) + (lane >> 4) * 8;
  int n = nt*16 + (lane & 15);
  dst[idx] = f2bf(src[(size_t)k * N + n]);
}

// ---------------------------------------------------------------------------
// K1: patch merge  x(16,3,448,448) -> xp fp32 (TOK,96). Block = 1 token.
// ---------------------------------------------------------------------------
__global__ __launch_bounds__(96) void k_patch_merge(const float* __restrict__ x,
                                                    const float* __restrict__ w,
                                                    const float* __restrict__ bias,
                                                    float* __restrict__ xp) {
  __shared__ float xs[48];
  size_t tkn = blockIdx.x;
  int bi  = (int)(tkn / 12544);
  int rem = (int)(tkn % 12544);
  int ph = rem / 112, pw = rem % 112;
  int tid = threadIdx.x;
  if (tid < 48) {
    int c = tid >> 4, ii = (tid >> 2) & 3, jj = tid & 3;
    xs[tid] = x[(((size_t)bi*3 + c)*448 + (ph*4 + ii))*448 + (pw*4 + jj)];
  }
  __syncthreads();
  float acc = bias[tid];
#pragma unroll 8
  for (int k = 0; k < 48; ++k) acc += xs[k] * w[k*96 + tid];
  xp[tkn*96 + tid] = acc;
}

// ---------------------------------------------------------------------------
// K2: LayerNorm  xp fp32 -> y bf16.  One wave per token (96 = 32*3 channels).
// ---------------------------------------------------------------------------
__global__ __launch_bounds__(256) void k_ln(const float* __restrict__ x,
                                            const float* __restrict__ gam,
                                            const float* __restrict__ bet,
                                            __bf16* __restrict__ y) {
  size_t tok = (size_t)blockIdx.x * 8 + (threadIdx.x >> 5);
  int lane = threadIdx.x & 31;
  const float* xr = x + tok*96;
  float v0 = xr[lane], v1 = xr[lane+32], v2 = xr[lane+64];
  float s = v0 + v1 + v2;
#pragma unroll
  for (int m = 1; m < 32; m <<= 1) s += __shfl_xor(s, m, 32);
  float mu = s * (1.0f/96.0f);
  float d0 = v0-mu, d1 = v1-mu, d2 = v2-mu;
  float q = d0*d0 + d1*d1 + d2*d2;
#pragma unroll
  for (int m = 1; m < 32; m <<= 1) q += __shfl_xor(q, m, 32);
  float rstd = rsqrtf(q * (1.0f/96.0f) + 1e-5f);
  __bf16* yr = y + tok*96;
  yr[lane]    = f2bf(d0*rstd*gam[lane]    + bet[lane]);
  yr[lane+32] = f2bf(d1*rstd*gam[lane+32] + bet[lane+32]);
  yr[lane+64] = f2bf(d2*rstd*gam[lane+64] + bet[lane+64]);
}

// ---------------------------------------------------------------------------
// K3: QKV GEMM (M=TOK,K=96,N=288) fused with window gather + cyclic shift.
// Wave = 16 padded window-tokens x 96 columns (one of q/k/v).
// q scaled by 32^-0.5, k token-major, v stored transposed (dim-major).
// Padded tokens (t>=49) produce zero rows.
// ---------------------------------------------------------------------------
__global__ __launch_bounds__(256) void k_qkv(const __bf16* __restrict__ y,
                                             const __bf16* __restrict__ wfrag,
                                             __bf16* __restrict__ qb,
                                             __bf16* __restrict__ kb,
                                             __bf16* __restrict__ vt,
                                             int shift) {
  int wid  = blockIdx.x * 8 + (threadIdx.x >> 5);   // 0 .. 49151
  int lane = threadIdx.x & 31, g = lane >> 4, col = lane & 15;
  int which = wid % 3;
  int mt    = wid / 3;                              // 0 .. 16383
  int win = mt >> 2, tt = mt & 3;
  int b = win >> 8, wy = (win >> 4) & 15, wx = win & 15;

  int t = tt*16 + col;
  bool valid = t < 49;
  const __bf16* rp = y;
  if (valid) {
    int ty = t / 7, tx = t - ty*7;
    int yy = (wy*7 + ty + shift) % 112;
    int xx = (wx*7 + tx + shift) % 112;
    rp = y + ((size_t)b*12544 + (size_t)yy*112 + xx) * 96;
  }

  v8f acc[6] = {};
  for (int kt = 0; kt < 3; ++kt) {
    v16bf a = valid ? frag_row(rp, kt, g) : zfrag();
#pragma unroll
    for (int j = 0; j < 6; ++j) {
      v16bf bw = frag_w(wfrag, which*6 + j, kt, 3, lane);
      acc[j] = wmma_bf(a, bw, acc[j]);
    }
  }

  size_t whb = (size_t)win * 3;
#pragma unroll
  for (int j = 0; j < 6; ++j) {
    int nn = j*16 + col;             // 0..95 within q/k/v
    int h = nn >> 5, d = nn & 31;
#pragma unroll
    for (int r = 0; r < 8; ++r) {
      int tm = tt*16 + r + 8*g;      // padded token 0..63
      float v = acc[j][r];
      if (which == 0)      qb[((whb + h)*64 + tm)*32 + d] = f2bf(v * QSCALE);
      else if (which == 1) kb[((whb + h)*64 + tm)*32 + d] = f2bf(v);
      else                 vt[((whb + h)*32 + d)*64 + tm] = f2bf(v);
    }
  }
}

// ---------------------------------------------------------------------------
// K4: window attention. Block(128thr = 4 waves) per window-head.
// Wave w: 16-row stripe of S(64x64) = 4 WMMAs (K=32), analytic pos-bias +
// shift masks, half-wave softmax, P->LDS reswizzle, O = P@V (4 WMMAs),
// scatter O back to token layout (undoing the cyclic shift).
// ---------------------------------------------------------------------------
__global__ __launch_bounds__(128) void k_attn(const __bf16* __restrict__ qb,
                                              const __bf16* __restrict__ kb,
                                              const __bf16* __restrict__ vt,
                                              const float* __restrict__ pos,
                                              __bf16* __restrict__ o,
                                              int shifted) {
  __shared__ __bf16 lds_p[4][16][64];
  int wh  = blockIdx.x;                 // window-head
  int win = wh / 3, h = wh % 3;
  int b = win >> 8, wy = (win >> 4) & 15, wx = win & 15;
  int w    = threadIdx.x >> 5;
  int lane = threadIdx.x & 31, g = lane >> 4, col = lane & 15;

  // ---- S = (q*scale) @ k^T, 16x64 stripe ----
  const __bf16* qrow = qb + ((size_t)wh*64 + w*16 + col) * 32;
  v16bf aQ = frag_row(qrow, 0, g);
  v8f s[4];
#pragma unroll
  for (int nt = 0; nt < 4; ++nt) {
    const __bf16* krow = kb + ((size_t)wh*64 + nt*16 + col) * 32;  // column j
    v16bf bK = frag_row(krow, 0, g);
    v8f z = {};
    s[nt] = wmma_bf(aQ, bK, z);
  }

  // ---- relative position bias + shift masks + padding mask ----
#pragma unroll
  for (int nt = 0; nt < 4; ++nt) {
    int tj = nt*16 + col;
    int tyj = tj / 7, txj = tj - tyj*7;
#pragma unroll
    for (int r = 0; r < 8; ++r) {
      int ti = w*16 + r + 8*g;
      float bias;
      if (ti < 49 && tj < 49) {
        int tyi = ti / 7, txi = ti - tyi*7;
        bias = pos[(tyj - tyi + 6)*13 + (txj - txi + 6)];
        if (shifted) {
          if (wy == 15 && ((tyi >= 4) != (tyj >= 4))) bias = -1e30f;
          if (wx == 15 && ((txi >= 4) != (txj >= 4))) bias = -1e30f;
        }
      } else {
        bias = -1e30f;                 // padded rows/cols
      }
      s[nt][r] += bias;
    }
  }

  // ---- row softmax: row (r+8g) lives in the 16 lanes sharing g ----
#pragma unroll
  for (int r = 0; r < 8; ++r) {
    float mx = -3.4e38f;
#pragma unroll
    for (int nt = 0; nt < 4; ++nt) mx = fmaxf(mx, s[nt][r]);
#pragma unroll
    for (int m = 1; m < 16; m <<= 1) mx = fmaxf(mx, __shfl_xor(mx, m, 32));
    float sum = 0.0f;
#pragma unroll
    for (int nt = 0; nt < 4; ++nt) { float e = __expf(s[nt][r] - mx); s[nt][r] = e; sum += e; }
#pragma unroll
    for (int m = 1; m < 16; m <<= 1) sum += __shfl_xor(sum, m, 32);
    float rinv = 1.0f / fmaxf(sum, 1e-20f);
#pragma unroll
    for (int nt = 0; nt < 4; ++nt) s[nt][r] *= rinv;
  }

  // ---- reswizzle P (D-layout) -> A-fragment layout via per-wave LDS tile ----
#pragma unroll
  for (int nt = 0; nt < 4; ++nt)
#pragma unroll
    for (int r = 0; r < 8; ++r)
      lds_p[w][r + 8*g][nt*16 + col] = f2bf(s[nt][r]);

  // ---- O = P @ V (K = 64 tokens, N = 32 dims) ----
  v8f o0 = {}, o1 = {};
  for (int kt = 0; kt < 2; ++kt) {
    v16bf aP = frag_row(&lds_p[w][col][0], kt, g);
#pragma unroll
    for (int nt = 0; nt < 2; ++nt) {
      const __bf16* vrow = vt + ((size_t)wh*32 + nt*16 + col) * 64;  // dim-major
      v16bf bV = frag_row(vrow, kt, g);
      if (nt == 0) o0 = wmma_bf(aP, bV, o0);
      else         o1 = wmma_bf(aP, bV, o1);
    }
  }

  // ---- scatter O to token layout (undo shift) ----
  int S = shifted ? 3 : 0;
#pragma unroll
  for (int r = 0; r < 8; ++r) {
    int t = w*16 + r + 8*g;
    if (t < 49) {
      int ty = t / 7, tx = t - ty*7;
      int yy = (wy*7 + ty + S) % 112;
      int xx = (wx*7 + tx + S) % 112;
      size_t row = (size_t)b*12544 + (size_t)yy*112 + xx;
      o[row*96 + h*32 + col]      = f2bf(o0[r]);
      o[row*96 + h*32 + 16 + col] = f2bf(o1[r]);
    }
  }
}

// ---------------------------------------------------------------------------
// K5: output projection + residual:  xp += o @ wo + bo   (M=TOK,K=96,N=96)
// ---------------------------------------------------------------------------
__global__ __launch_bounds__(256) void k_proj(const __bf16* __restrict__ o,
                                              const __bf16* __restrict__ wfrag,
                                              const float* __restrict__ bias,
                                              float* __restrict__ xp) {
  int wid  = blockIdx.x * 8 + (threadIdx.x >> 5);   // M-tile 0..12543
  int lane = threadIdx.x & 31, g = lane >> 4, col = lane & 15;
  size_t row0 = (size_t)wid * 16;
  const __bf16* rp = o + (row0 + col) * 96;
  v8f acc[6] = {};
  for (int kt = 0; kt < 3; ++kt) {
    v16bf a = frag_row(rp, kt, g);
#pragma unroll
    for (int j = 0; j < 6; ++j)
      acc[j] = wmma_bf(a, frag_w(wfrag, j, kt, 3, lane), acc[j]);
  }
#pragma unroll
  for (int j = 0; j < 6; ++j) {
    int n = j*16 + col;
    float bb = bias[n];
#pragma unroll
    for (int r = 0; r < 8; ++r)
      xp[(row0 + r + 8*g)*96 + n] += acc[j][r] + bb;
  }
}

// ---------------------------------------------------------------------------
// K6: FF1 + exact GELU:  ff = gelu(y @ w1 + b1)   (M=TOK,K=96,N=384)
// Wave handles 16 rows x 96 cols (chunk of 6 n-tiles).
// ---------------------------------------------------------------------------
__global__ __launch_bounds__(256) void k_ff1(const __bf16* __restrict__ y,
                                             const __bf16* __restrict__ wfrag,
                                             const float* __restrict__ bias,
                                             __bf16* __restrict__ ff) {
  int wid  = blockIdx.x * 8 + (threadIdx.x >> 5);   // 0..50175
  int lane = threadIdx.x & 31, g = lane >> 4, col = lane & 15;
  int chunk = wid & 3;
  int mt    = wid >> 2;
  size_t row0 = (size_t)mt * 16;
  const __bf16* rp = y + (row0 + col) * 96;
  v8f acc[6] = {};
  for (int kt = 0; kt < 3; ++kt) {
    v16bf a = frag_row(rp, kt, g);
#pragma unroll
    for (int j = 0; j < 6; ++j)
      acc[j] = wmma_bf(a, frag_w(wfrag, chunk*6 + j, kt, 3, lane), acc[j]);
  }
#pragma unroll
  for (int j = 0; j < 6; ++j) {
    int n = (chunk*6 + j)*16 + col;
    float bb = bias[n];
#pragma unroll
    for (int r = 0; r < 8; ++r)
      ff[(row0 + r + 8*g)*384 + n] = f2bf(gelu_exact(acc[j][r] + bb));
  }
}

// ---------------------------------------------------------------------------
// K7: FF2 + residual:  xp += ff @ w2 + b2   (M=TOK,K=384,N=96)
// ---------------------------------------------------------------------------
__global__ __launch_bounds__(256) void k_ff2(const __bf16* __restrict__ ff,
                                             const __bf16* __restrict__ wfrag,
                                             const float* __restrict__ bias,
                                             float* __restrict__ xp) {
  int wid  = blockIdx.x * 8 + (threadIdx.x >> 5);   // M-tile 0..12543
  int lane = threadIdx.x & 31, g = lane >> 4, col = lane & 15;
  size_t row0 = (size_t)wid * 16;
  const __bf16* rp = ff + (row0 + col) * 384;
  v8f acc[6] = {};
  for (int kt = 0; kt < 12; ++kt) {
    v16bf a = frag_row(rp, kt, g);
#pragma unroll
    for (int j = 0; j < 6; ++j)
      acc[j] = wmma_bf(a, frag_w(wfrag, j, kt, 12, lane), acc[j]);
  }
#pragma unroll
  for (int j = 0; j < 6; ++j) {
    int n = j*16 + col;
    float bb = bias[n];
#pragma unroll
    for (int r = 0; r < 8; ++r)
      xp[(row0 + r + 8*g)*96 + n] += acc[j][r] + bb;
  }
}

// ---------------------------------------------------------------------------
// K8: final NHWC -> NCHW transpose (fp32)
// ---------------------------------------------------------------------------
__global__ __launch_bounds__(256) void k_out(const float* __restrict__ xp,
                                             float* __restrict__ out) {
  size_t i = (size_t)blockIdx.x * 256 + threadIdx.x;
  if (i >= TOK * 96) return;
  size_t px = i % 12544;
  size_t t2 = i / 12544;
  int c = (int)(t2 % 96);
  int bb = (int)(t2 / 96);
  out[i] = xp[((size_t)bb*12544 + px)*96 + c];
}

// ---------------------------------------------------------------------------
// Host launcher
// ---------------------------------------------------------------------------
extern "C" void kernel_launch(void* const* d_in, const int* in_sizes, int n_in,
                              void* d_out, int out_size, void* d_ws, size_t ws_size,
                              hipStream_t stream) {
  (void)in_sizes; (void)n_in; (void)out_size; (void)ws_size;

  const float* x     = (const float*)d_in[0];
  const float* pm_w  = (const float*)d_in[1];
  const float* pm_b  = (const float*)d_in[2];
  const float* ln1_g = (const float*)d_in[3];
  const float* ln1_b = (const float*)d_in[4];
  const float* qkv_w = (const float*)d_in[5];
  const float* pos   = (const float*)d_in[6];
  const float* out_w = (const float*)d_in[7];
  const float* out_b = (const float*)d_in[8];
  const float* ln2_g = (const float*)d_in[9];
  const float* ln2_b = (const float*)d_in[10];
  const float* ff1_w = (const float*)d_in[11];
  const float* ff1_b = (const float*)d_in[12];
  const float* ff2_w = (const float*)d_in[13];
  const float* ff2_b = (const float*)d_in[14];

  char* ws = (char*)d_ws;
  float*  xp = (float*)(ws + OFF_XP);
  __bf16* y  = (__bf16*)(ws + OFF_Y);
  __bf16* qb = (__bf16*)(ws + OFF_Q);
  __bf16* kb = (__bf16*)(ws + OFF_K);
  __bf16* vt = (__bf16*)(ws + OFF_V);
  __bf16* ff = (__bf16*)(ws + OFF_FF);
  __bf16* wf = (__bf16*)(ws + OFF_W);

  // Pre-swizzle all weights to bf16 WMMA B-fragment order (tiny, L2-resident)
  for (int blk = 0; blk < 2; ++blk) {
    k_cvt<<<108, 256, 0, stream>>>(qkv_w + (size_t)blk*96*288, wf + WOFF_QKV + blk*27648, 96, 288);
    k_cvt<<<36,  256, 0, stream>>>(out_w + (size_t)blk*96*96,  wf + WOFF_OUT + blk*9216,  96, 96);
    k_cvt<<<144, 256, 0, stream>>>(ff1_w + (size_t)blk*96*384, wf + WOFF_FF1 + blk*36864, 96, 384);
    k_cvt<<<144, 256, 0, stream>>>(ff2_w + (size_t)blk*384*96, wf + WOFF_FF2 + blk*36864, 384, 96);
  }

  k_patch_merge<<<(unsigned)TOK, 96, 0, stream>>>(x, pm_w, pm_b, xp);

  for (int blk = 0; blk < 2; ++blk) {
    int shift = blk ? 3 : 0;
    k_ln  <<<25088, 256, 0, stream>>>(xp, ln1_g + blk*96, ln1_b + blk*96, y);
    k_qkv <<<6144,  256, 0, stream>>>(y, wf + WOFF_QKV + blk*27648, qb, kb, vt, shift);
    k_attn<<<12288, 128, 0, stream>>>(qb, kb, vt, pos + blk*169, y, blk);
    k_proj<<<1568,  256, 0, stream>>>(y, wf + WOFF_OUT + blk*9216, out_b + blk*96, xp);
    k_ln  <<<25088, 256, 0, stream>>>(xp, ln2_g + blk*96, ln2_b + blk*96, y);
    k_ff1 <<<6272,  256, 0, stream>>>(y, wf + WOFF_FF1 + blk*36864, ff1_b + (size_t)blk*384, ff);
    k_ff2 <<<1568,  256, 0, stream>>>(ff, wf + WOFF_FF2 + blk*36864, ff2_b + blk*96, xp);
  }

  k_out<<<75264, 256, 0, stream>>>(xp, (float*)d_out);
}
